// SA_32040456028259
// MI455X (gfx1250) — compile-verified, matrix operand order checked by reading, not run on hardware
//
#include <hip/hip_runtime.h>
#include <hip/hip_bf16.h>

// Shapes from the reference
#define Bsz 512
#define Tt  64
#define Dd  128
#define Hh  128
#define BB  16        // batch rows per block == one WMMA M-tile
#define NTHREADS 1024
#define NWAVES   32

typedef __attribute__((ext_vector_type(16))) __bf16 v16bf;
typedef __attribute__((ext_vector_type(8)))  float  v8f;

__device__ __forceinline__ float fast_tanh(float x) {
    // tanh(x) = 1 - 2/(1+e^{2x});  e^{2x}->inf => 1, ->0 => -1 (saturates cleanly)
    float e = __expf(2.0f * x);
    return 1.0f - 2.0f * __builtin_amdgcn_rcpf(e + 1.0f);
}
__device__ __forceinline__ float fast_sig(float x) {
    return __builtin_amdgcn_rcpf(1.0f + __expf(-x));
}

// ISA 7.12.2: 16-bit A 16x32 — VGPR p holds K pair base:
__device__ __forceinline__ int kOfPair(int p, int hl) {
    return ((p & 4) ? 16 : 0) + (p & 3) * 2 + hl * 8;
}

__device__ __forceinline__ v8f wmma_bf16(v16bf a, v16bf b, v8f c) {
    return __builtin_amdgcn_wmma_f32_16x16x32_bf16(
        false, a, false, b, (short)0, c, false, false);
}

// ---------------- Kernel A: attribute attention alpha [D] -------------------
__global__ void SA_attr_alpha(const float* __restrict__ ad,
                              const float* __restrict__ w1,
                              const float* __restrict__ b1,
                              const float* __restrict__ w2,
                              const float* __restrict__ b2,
                              float* __restrict__ alpha) {
    __shared__ float se[Dd];
    int d = threadIdx.x;                 // 128 threads
    float e = b2[0];
    for (int j = 0; j < 32; ++j) {
        float a = b1[j];
        for (int k = 0; k < 64; ++k) a += ad[d * 64 + k] * w1[k * 32 + j];
        a = a > 0.0f ? a : 0.5f * a;     // LeakyReLU(0.5)
        e += a * w2[j];
    }
    se[d] = e;
    __syncthreads();
    float m = -1e30f;
    for (int i = 0; i < Dd; ++i) m = fmaxf(m, se[i]);
    float s = 0.0f;
    for (int i = 0; i < Dd; ++i) s += __expf(se[i] - m);
    alpha[d] = __expf(e - m) / s;
}

// ---------------- Kernel B: full recurrence, 16 batch rows / block ----------
__global__ void __launch_bounds__(NTHREADS, 1)
SA_darnn(const float* __restrict__ X,
         const float* __restrict__ w_a1, const float* __restrict__ b_a1_g,
         const float* __restrict__ w_a2_g, const float* __restrict__ b_a2_g,
         const float* __restrict__ w_ih, const float* __restrict__ w_hh,
         const float* __restrict__ b_ih, const float* __restrict__ b_hh,
         const float* __restrict__ alphaA, float* __restrict__ P,
         float* __restrict__ out) {
    __shared__ __bf16 sAB[BB][256 + 8];   // [h|c] bf16 (GEMM1 A), padded
    __shared__ __bf16 sXH[BB][256 + 8];   // [x_t|h] bf16 (GEMM3 A), padded
    __shared__ float  sXt[BB][Dd];        // x_t (f32, for output)
    __shared__ float  sHC[BB][Hh + 4];    // hcproj
    __shared__ float  sE[BB][Dd];         // e scores
    __shared__ float  sG[BB][4 * Hh + 4]; // gates
    __shared__ float  sH[BB][Hh], sC[BB][Hh];
    __shared__ float  sBa1[Hh], sWa2[Hh], sBias[4 * Hh], sAl[Dd];
    __shared__ float  sBa2;

    const int tid  = threadIdx.x;
    const int lane = tid & 31;
    const int wv   = tid >> 5;           // 0..31
    const int lm   = lane & 15;          // M/N index in tile
    const int hl   = lane >> 4;          // lane half
    const int gb0  = blockIdx.x * BB;

    for (int i = tid; i < Hh; i += NTHREADS) { sBa1[i] = b_a1_g[i]; sWa2[i] = w_a2_g[i]; }
    for (int i = tid; i < 4 * Hh; i += NTHREADS) sBias[i] = b_ih[i] + b_hh[i];
    for (int i = tid; i < Dd; i += NTHREADS) sAl[i] = alphaA[i];
    if (tid == 0) sBa2 = b_a2_g[0];
    for (int i = tid; i < BB * (256 + 8); i += NTHREADS) {
        ((__bf16*)sAB)[i] = (__bf16)0.0f;
        ((__bf16*)sXH)[i] = (__bf16)0.0f;
    }
    for (int i = tid; i < BB * Hh; i += NTHREADS) { ((float*)sH)[i] = 0.0f; ((float*)sC)[i] = 0.0f; }
    __syncthreads();

    // ---- Prologue: P[b][h][d] = sum_t (X[b][t][d]*alpha[d]) * w_a1[256+t][h]
    // Batched [128x64]@[64x128] via bf16 WMMA. 16 b * 8 mt * 8 nt jobs.
    for (int job = wv; job < 16 * 64; job += NWAVES) {
        const int b  = job >> 6;
        const int mt = (job >> 3) & 7;   // d tile
        const int nt = job & 7;          // h tile
        const int gb = gb0 + b;
        v8f acc = {};
        #pragma unroll
        for (int kt = 0; kt < 2; ++kt) {
            v16bf af, bf;
            const int d = mt * 16 + lm;
            const float al = sAl[d];
            const int n = nt * 16 + lm;
            #pragma unroll
            for (int p = 0; p < 8; ++p) {
                const int kb = kt * 32 + kOfPair(p, hl);     // = t
                af[2 * p]     = (__bf16)(X[((size_t)gb * Tt + kb) * Dd + d] * al);
                af[2 * p + 1] = (__bf16)(X[((size_t)gb * Tt + kb + 1) * Dd + d] * al);
                bf[2 * p]     = (__bf16)w_a1[(256 + kb) * Hh + n];
                bf[2 * p + 1] = (__bf16)w_a1[(256 + kb + 1) * Hh + n];
            }
            acc = wmma_bf16(af, bf, acc);
        }
        #pragma unroll
        for (int r = 0; r < 8; ++r) {
            const int d = mt * 16 + r + 8 * hl;
            const int h = nt * 16 + lm;
            P[((size_t)gb * Hh + h) * Dd + d] = acc[r];      // transposed for e-stage coalescing
        }
    }
    __threadfence();
    __syncthreads();

    // ---- Persistent B fragments in VGPRs: ONLY the gates GEMM (biggest reuse,
    // ~1 GB of L2 traffic saved over the run). 1 n-tile/wave = 64 VGPRs, which
    // fits the per-wave budget at 8 waves/SIMD without spilling.
    v16bf Bg[8];
    {
        const int n = wv * 16 + lm;
        #pragma unroll
        for (int kt = 0; kt < 8; ++kt) {
            v16bf f;
            #pragma unroll
            for (int p = 0; p < 8; ++p) {
                const int k = kt * 32 + kOfPair(p, hl);
                float v0, v1;
                if (k < 128) { v0 = w_ih[n * Dd + k];       v1 = w_ih[n * Dd + k + 1]; }
                else         { v0 = w_hh[n * Dd + k - 128]; v1 = w_hh[n * Dd + k - 127]; }
                f[2 * p] = (__bf16)v0; f[2 * p + 1] = (__bf16)v1;
            }
            Bg[kt] = f;
        }
    }

    // ---------------- sequential scan over t ----------------
    for (int t = 0; t < Tt; ++t) {
        // (1) load x_t = X[:,t,:] * alpha
        for (int i = tid; i < BB * Dd; i += NTHREADS) {
            const int bb = i >> 7, d = i & 127;
            const float v = X[((size_t)(gb0 + bb) * Tt + t) * Dd + d] * sAl[d];
            sXt[bb][d] = v;
            sXH[bb][d] = (__bf16)v;
        }
        __syncthreads();

        // (2) hcproj = [h|c] @ w_a1[0:256]   (waves 0-7, WMMA)
        // B fragments streamed from global each step: w_a1 is 160 KB, fully
        // L2-resident; 128 KB/block/step => ~268 MB total. Cheap, and keeps
        // persistent VGPR usage at 64/wave (no spills).
        if (wv < 8) {
            const int n = wv * 16 + lm;
            v8f acc = {};
            #pragma unroll
            for (int kt = 0; kt < 8; ++kt) {
                v16bf a, bfr;
                #pragma unroll
                for (int p = 0; p < 8; ++p) {
                    const int kb = kt * 32 + kOfPair(p, hl);
                    unsigned int u = *(const unsigned int*)&sAB[lm][kb];
                    a[2 * p]     = ((__bf16*)&u)[0];
                    a[2 * p + 1] = ((__bf16*)&u)[1];
                    bfr[2 * p]     = (__bf16)w_a1[kb * Hh + n];
                    bfr[2 * p + 1] = (__bf16)w_a1[(kb + 1) * Hh + n];
                }
                acc = wmma_bf16(a, bfr, acc);
            }
            #pragma unroll
            for (int r = 0; r < 8; ++r) sHC[r + 8 * hl][wv * 16 + lm] = acc[r];
        }
        __syncthreads();

        // (3) e[bb][d] = sum_h tanh(P + hcproj + b_a1) * w_a2 + b_a2
        for (int i = tid; i < BB * Dd; i += NTHREADS) {   // 2 iterations
            const int bb = i >> 7, d = i & 127;
            const float* Pr = P + (size_t)(gb0 + bb) * Hh * Dd + d;
            float acc = sBa2;
            #pragma unroll 4
            for (int h = 0; h < Hh; ++h) {
                acc += fast_tanh(Pr[h * Dd] + sHC[bb][h] + sBa1[h]) * sWa2[h];
            }
            sE[bb][d] = acc;
        }
        __syncthreads();

        // (4a) softmax over d + write x_tilde   (waves 0-15, one row each)
        if (wv < BB) {
            const int bb = wv;
            float e0 = sE[bb][lane],      e1 = sE[bb][lane + 32];
            float e2 = sE[bb][lane + 64], e3 = sE[bb][lane + 96];
            float m = fmaxf(fmaxf(e0, e1), fmaxf(e2, e3));
            for (int off = 16; off > 0; off >>= 1) m = fmaxf(m, __shfl_xor(m, off, 32));
            float x0 = __expf(e0 - m), x1 = __expf(e1 - m);
            float x2 = __expf(e2 - m), x3 = __expf(e3 - m);
            float s = x0 + x1 + x2 + x3;
            for (int off = 16; off > 0; off >>= 1) s += __shfl_xor(s, off, 32);
            const float rs = __builtin_amdgcn_rcpf(s);
            const size_t ob = ((size_t)(gb0 + bb) * Tt + t) * Dd;
            out[ob + lane]      = x0 * rs * sXt[bb][lane];
            out[ob + lane + 32] = x1 * rs * sXt[bb][lane + 32];
            out[ob + lane + 64] = x2 * rs * sXt[bb][lane + 64];
            out[ob + lane + 96] = x3 * rs * sXt[bb][lane + 96];
        }

        // (4b) gates = [x_t|h] @ [w_ih;w_hh]^T   (all 32 waves, WMMA,
        //      B operand register-resident for the whole scan)
        {
            v8f acc = {};
            #pragma unroll
            for (int kt = 0; kt < 8; ++kt) {
                v16bf a;
                #pragma unroll
                for (int p = 0; p < 8; ++p) {
                    const int kb = kt * 32 + kOfPair(p, hl);
                    unsigned int u = *(const unsigned int*)&sXH[lm][kb];
                    a[2 * p]     = ((__bf16*)&u)[0];
                    a[2 * p + 1] = ((__bf16*)&u)[1];
                }
                acc = wmma_bf16(a, Bg[kt], acc);
            }
            #pragma unroll
            for (int r = 0; r < 8; ++r) sG[r + 8 * hl][wv * 16 + lm] = acc[r];
        }
        __syncthreads();

        // (5) LSTM cell update (waves 0-15, one row each)
        if (wv < BB) {
            const int bb = wv;
            #pragma unroll
            for (int k = 0; k < 4; ++k) {
                const int h = lane + 32 * k;
                const float ig = sG[bb][h]           + sBias[h];
                const float fg = sG[bb][Hh + h]      + sBias[Hh + h];
                const float gg = sG[bb][2 * Hh + h]  + sBias[2 * Hh + h];
                const float og = sG[bb][3 * Hh + h]  + sBias[3 * Hh + h];
                const float c  = fast_sig(fg) * sC[bb][h] + fast_sig(ig) * fast_tanh(gg);
                const float hn = fast_sig(og) * fast_tanh(c);
                sC[bb][h] = c;  sH[bb][h] = hn;
                sAB[bb][h]        = (__bf16)hn;   // GEMM1 A for next step
                sAB[bb][128 + h]  = (__bf16)c;
                sXH[bb][128 + h]  = (__bf16)hn;   // GEMM3 A for next step
            }
        }
        __syncthreads();
    }
}

extern "C" void kernel_launch(void* const* d_in, const int* in_sizes, int n_in,
                              void* d_out, int out_size, void* d_ws, size_t ws_size,
                              hipStream_t stream) {
    const float* X       = (const float*)d_in[0];
    const float* attri   = (const float*)d_in[1];
    const float* w_attr1 = (const float*)d_in[2];
    const float* b_attr1 = (const float*)d_in[3];
    const float* w_attr2 = (const float*)d_in[4];
    const float* b_attr2 = (const float*)d_in[5];
    const float* w_a1    = (const float*)d_in[6];
    const float* b_a1    = (const float*)d_in[7];
    const float* w_a2    = (const float*)d_in[8];
    const float* b_a2    = (const float*)d_in[9];
    const float* w_ih    = (const float*)d_in[10];
    const float* w_hh    = (const float*)d_in[11];
    const float* b_ih    = (const float*)d_in[12];
    const float* b_hh    = (const float*)d_in[13];
    float* out = (float*)d_out;

    float* ws    = (float*)d_ws;
    float* alpha = ws;           // 128 floats
    float* P     = ws + 128;     // 512*128*128 floats = 33.5 MB (L2-resident)

    SA_attr_alpha<<<1, 128, 0, stream>>>(attri, w_attr1, b_attr1, w_attr2, b_attr2, alpha);
    SA_darnn<<<Bsz / BB, NTHREADS, 0, stream>>>(X, w_a1, b_a1, w_a2, b_a2,
                                                w_ih, w_hh, b_ih, b_hh,
                                                alpha, P, out);
}
